// KPNet_27642409517712
// MI455X (gfx1250) — matrix operand order repaired and checked by reading, not compiled, not run on hardware
//
#include <hip/hip_runtime.h>
#include <hip/hip_fp16.h>
#include <hip/hip_bf16.h>

// ---------------- problem constants ----------------
#define BDIM 4
#define NPT  16384
#define SPT  512
#define KNB  64
#define GKNB 20
#define DILK 128            // K*DIL
#define M1   (BDIM*SPT*KNB) // 131072 cluster points
#define M2   (BDIM*SPT*GKNB)// 40960 edge points
#define PINF 3.0e38f

typedef _Float16 h16;
typedef __attribute__((ext_vector_type(16))) _Float16 v16h;
typedef __attribute__((ext_vector_type(8)))  _Float16 v8h;
typedef __attribute__((ext_vector_type(8)))  float    v8f;

__device__ __forceinline__ float apply_act(float x, int act) {
  if (act == 1) return fmaxf(x, 0.0f);
  if (act == 2) return x > 0.0f ? x : 0.2f * x;
  return x;
}

__device__ __forceinline__ v16h cat8(v8h lo, v8h hi8) {
  v16h r;
#pragma unroll
  for (int e = 0; e < 8; ++e) { r[e] = lo[e]; r[8 + e] = hi8[e]; }
  return r;
}

// ===== convert (and zero-pad) f32 weights [N,K] -> h16 [N,Kpad] =====
__global__ void cvt_weights_kernel(const float* __restrict__ W, h16* __restrict__ Wp,
                                   int N, int K, int Kpad)
{
  int t = blockIdx.x * blockDim.x + threadIdx.x;
  if (t >= N * Kpad) return;
  int n = t / Kpad, k = t % Kpad;
  Wp[t] = (k < K) ? (h16)W[(size_t)n * K + k] : (h16)0.0f;
}

// ================= WMMA GEMM: Y = act(bn(X[M,K] @ W[N,K]^T (+bias))) =========
// All-h16 data path. CONTRACT: M%16==0, K%32==0, N%(16*NT)==0.
// One wave computes a 16 x (16*NT) strip: A fragment loaded once, NT WMMAs.
template <int NT>
__global__ __launch_bounds__(32)
void gemm_h16_nt(const h16* __restrict__ X, const h16* __restrict__ W,
                 const float* __restrict__ bias,
                 const float* __restrict__ gamma, const float* __restrict__ beta,
                 h16* __restrict__ Y, int M, int K, int N, int act)
{
  const int lane = threadIdx.x;
  const int lr = lane & 15;
  const int hi = lane >> 4;
  const int m0 = blockIdx.x * 16;
  const int n0 = blockIdx.y * (16 * NT);
  const h16* xrow = X + (size_t)(m0 + lr) * K + (hi << 4);
  const h16* wrow[NT];
#pragma unroll
  for (int j = 0; j < NT; ++j)
    wrow[j] = W + (size_t)(n0 + j * 16 + lr) * K + (hi << 4);
  v8f acc[NT];
#pragma unroll
  for (int j = 0; j < NT; ++j) acc[j] = (v8f){0.f,0.f,0.f,0.f,0.f,0.f,0.f,0.f};

  for (int kb = 0; kb < K; kb += 32) {
    // A: 16 contiguous halfs per lane (lanes0-15 k=0..15, lanes16-31 k=16..31)
    v8h x0 = *(const v8h*)(xrow + kb);
    v8h x1 = *(const v8h*)(xrow + kb + 8);
    v16h af = cat8(x0, x1);
#pragma unroll
    for (int j = 0; j < NT; ++j) {
      // B: 16 contiguous halfs per lane (lanes0-15 k=0..15, lanes16-31 k=16..31)
      v16h bf = *(const v16h*)(wrow[j] + kb);
      acc[j] = __builtin_amdgcn_wmma_f32_16x16x32_f16(false, af, false, bf, (short)0,
                                                      acc[j], false, false);
    }
  }
#pragma unroll
  for (int j = 0; j < NT; ++j) {
    const int ncol = n0 + j * 16 + lr;
    const float scale = gamma ? gamma[ncol] * rsqrtf(1.0f + 1e-5f) : 1.0f;
    const float shift = gamma ? beta[ncol] : 0.0f;
    const float bs = bias ? bias[ncol] : 0.0f;
#pragma unroll
    for (int r = 0; r < 8; ++r) {
      const int m = m0 + (hi << 3) + r;
      float v = (acc[j][r] + bs) * scale + shift;
      Y[(size_t)m * N + ncol] = (h16)apply_act(v, act);
    }
  }
}

// ===== fused y4 = relu(bn4( [ym(bcast,128) | y3(128) | e3*a(256)] @ s_w4^T )) =====
// Split accumulators: y4 = bn(acc1 + a[m]*acc2); no per-element scaling in loop.
__global__ __launch_bounds__(32)
void gemm_y4_kernel(const h16* __restrict__ ym, const h16* __restrict__ y3,
                    const h16* __restrict__ e3, const float* __restrict__ a,
                    const h16* __restrict__ W, const float* __restrict__ gamma,
                    const float* __restrict__ beta, h16* __restrict__ Y, int M)
{
  const int NT = 4;                  // N=128, grid.y == 2
  const int lane = threadIdx.x;
  const int lr = lane & 15;
  const int hi = lane >> 4;
  const int m0 = blockIdx.x * 16;
  const int n0 = blockIdx.y * (16 * NT);
  const int mrow = m0 + lr;
  const h16* wrow[NT];
#pragma unroll
  for (int j = 0; j < NT; ++j)
    wrow[j] = W + (size_t)(n0 + j * 16 + lr) * 512 + (hi << 4);
  v8f acc1[NT], acc2[NT];
#pragma unroll
  for (int j = 0; j < NT; ++j) {
    acc1[j] = (v8f){0.f,0.f,0.f,0.f,0.f,0.f,0.f,0.f};
    acc2[j] = (v8f){0.f,0.f,0.f,0.f,0.f,0.f,0.f,0.f};
  }
  // segment 1: ym (row-broadcast over the K=64 neighbors)
  const h16* s1 = ym + (size_t)(mrow >> 6) * 128 + (hi << 4);
  for (int kb = 0; kb < 128; kb += 32) {
    v16h af = cat8(*(const v8h*)(s1 + kb), *(const v8h*)(s1 + kb + 8));
#pragma unroll
    for (int j = 0; j < NT; ++j) {
      v16h bf = *(const v16h*)(wrow[j] + kb);
      acc1[j] = __builtin_amdgcn_wmma_f32_16x16x32_f16(false, af, false, bf, (short)0,
                                                       acc1[j], false, false);
    }
  }
  // segment 2: y3
  const h16* s2 = y3 + (size_t)mrow * 128 + (hi << 4);
  for (int kb = 0; kb < 128; kb += 32) {
    v16h af = cat8(*(const v8h*)(s2 + kb), *(const v8h*)(s2 + kb + 8));
#pragma unroll
    for (int j = 0; j < NT; ++j) {
      v16h bf = *(const v16h*)(wrow[j] + 128 + kb);
      acc1[j] = __builtin_amdgcn_wmma_f32_16x16x32_f16(false, af, false, bf, (short)0,
                                                       acc1[j], false, false);
    }
  }
  // segment 3: e3 (scaled by a[m] AFTER accumulation, exploiting linearity)
  const h16* s3 = e3 + (size_t)mrow * 256 + (hi << 4);
  for (int kb = 0; kb < 256; kb += 32) {
    v16h af = cat8(*(const v8h*)(s3 + kb), *(const v8h*)(s3 + kb + 8));
#pragma unroll
    for (int j = 0; j < NT; ++j) {
      v16h bf = *(const v16h*)(wrow[j] + 256 + kb);
      acc2[j] = __builtin_amdgcn_wmma_f32_16x16x32_f16(false, af, false, bf, (short)0,
                                                       acc2[j], false, false);
    }
  }
#pragma unroll
  for (int j = 0; j < NT; ++j) {
    const int ncol = n0 + j * 16 + lr;
    const float scale = gamma[ncol] * rsqrtf(1.0f + 1e-5f);
    const float shift = beta[ncol];
#pragma unroll
    for (int r = 0; r < 8; ++r) {
      const int m = m0 + (hi << 3) + r;
      float v = (acc1[j][r] + a[m] * acc2[j][r]) * scale + shift;
      Y[(size_t)m * 128 + ncol] = (h16)fmaxf(v, 0.0f);
    }
  }
}

// ===== top-128 neighbor search: distances held in LDS, iterative extract-min =====
__global__ void topk128_kernel(const float* __restrict__ xyz, const int* __restrict__ sidx,
                               int* __restrict__ out_idx)
{
  extern __shared__ float smemf[];
  float* dists = smemf;                         // NPT floats (64 KB)
  float* rmin  = smemf + NPT;
  int*   rarg  = (int*)(rmin + blockDim.x);
  const int b = blockIdx.x / SPT;
  const int s = blockIdx.x % SPT;
  const float* base = xyz + (size_t)b * NPT * 3;
  const int sp = sidx[s];
  const float cx = base[(size_t)sp * 3 + 0];
  const float cy = base[(size_t)sp * 3 + 1];
  const float cz = base[(size_t)sp * 3 + 2];
  for (int n = threadIdx.x; n < NPT; n += blockDim.x) {
    float dx = base[(size_t)n * 3 + 0] - cx;
    float dy = base[(size_t)n * 3 + 1] - cy;
    float dz = base[(size_t)n * 3 + 2] - cz;
    dists[n] = dx * dx + dy * dy + dz * dz;
  }
  __syncthreads();
  for (int it = 0; it < DILK; ++it) {
    float best = PINF; int barg = 0x7fffffff;
    for (int n = threadIdx.x; n < NPT; n += blockDim.x) {
      float v = dists[n];
      if (v < best || (v == best && n < barg)) { best = v; barg = n; }
    }
    rmin[threadIdx.x] = best; rarg[threadIdx.x] = barg;
    __syncthreads();
    for (int off = blockDim.x >> 1; off > 0; off >>= 1) {
      if ((int)threadIdx.x < off) {
        float v2 = rmin[threadIdx.x + off]; int a2 = rarg[threadIdx.x + off];
        if (v2 < rmin[threadIdx.x] ||
            (v2 == rmin[threadIdx.x] && a2 < rarg[threadIdx.x])) {
          rmin[threadIdx.x] = v2; rarg[threadIdx.x] = a2;
        }
      }
      __syncthreads();
    }
    if (threadIdx.x == 0) {
      int w = rarg[0];
      out_idx[(size_t)blockIdx.x * DILK + it] = w;
      dists[w] = PINF;
    }
    __syncthreads();
  }
}

// ===== knn (top-GK of P=512) over h16 features =====
__global__ void knn_kernel(const h16* __restrict__ feat, int ld, int C,
                           int P, int topk, int* __restrict__ outIdx)
{
  extern __shared__ float smemf[];
  float* d    = smemf;                          // P floats
  float* rmin = smemf + P;
  int*   rarg = (int*)(rmin + blockDim.x);
  const int b = blockIdx.x / P;
  const int i = blockIdx.x % P;
  const h16* fb = feat + (size_t)b * P * ld;
  for (int j = threadIdx.x; j < P; j += blockDim.x) {
    float s = 0.0f;
    for (int c = 0; c < C; ++c) {
      float dd = (float)fb[(size_t)i * ld + c] - (float)fb[(size_t)j * ld + c];
      s += dd * dd;
    }
    d[j] = s;
  }
  __syncthreads();
  for (int it = 0; it < topk; ++it) {
    float best = PINF; int barg = 0x7fffffff;
    for (int j = threadIdx.x; j < P; j += blockDim.x) {
      float v = d[j];
      if (v < best || (v == best && j < barg)) { best = v; barg = j; }
    }
    rmin[threadIdx.x] = best; rarg[threadIdx.x] = barg;
    __syncthreads();
    for (int off = blockDim.x >> 1; off > 0; off >>= 1) {
      if ((int)threadIdx.x < off) {
        float v2 = rmin[threadIdx.x + off]; int a2 = rarg[threadIdx.x + off];
        if (v2 < rmin[threadIdx.x] ||
            (v2 == rmin[threadIdx.x] && a2 < rarg[threadIdx.x])) {
          rmin[threadIdx.x] = v2; rarg[threadIdx.x] = a2;
        }
      }
      __syncthreads();
    }
    if (threadIdx.x == 0) {
      int w = rarg[0];
      outIdx[(size_t)blockIdx.x * topk + it] = w;
      d[w] = PINF;
    }
    __syncthreads();
  }
}

// ===== gather neighbors, build h16 cluster = [rel(3), dist, 0-pad(28)] =====
__global__ void gather_cluster_kernel(const float* __restrict__ xyz,
                                      const int* __restrict__ sidx,
                                      const int* __restrict__ dsel,
                                      const int* __restrict__ top128,
                                      float* __restrict__ nx, h16* __restrict__ cluster)
{
  int t = blockIdx.x * blockDim.x + threadIdx.x;
  if (t >= M1) return;
  const int k  = t % KNB;
  const int bs = t / KNB;
  const int s  = bs % SPT;
  const int b  = bs / SPT;
  const int g  = top128[(size_t)bs * DILK + dsel[k]];
  const int sp = sidx[s];
  const float* base = xyz + (size_t)b * NPT * 3;
  float cx = base[(size_t)sp * 3 + 0], cy = base[(size_t)sp * 3 + 1], cz = base[(size_t)sp * 3 + 2];
  float px = base[(size_t)g  * 3 + 0], py = base[(size_t)g  * 3 + 1], pz = base[(size_t)g  * 3 + 2];
  float rx = px - cx, ry = py - cy, rz = pz - cz;
  float dist = sqrtf(fmaxf(rx * rx + ry * ry + rz * rz, 1e-12f));
  nx[(size_t)t * 3 + 0] = px; nx[(size_t)t * 3 + 1] = py; nx[(size_t)t * 3 + 2] = pz;
  h16* cr = cluster + (size_t)t * 32;
  cr[0] = (h16)rx; cr[1] = (h16)ry; cr[2] = (h16)rz; cr[3] = (h16)dist;
#pragma unroll
  for (int c = 4; c < 32; ++c) cr[c] = (h16)0.0f;
}

// ===== channel-max of e3 per cluster point =====
__global__ void emax_kernel(const h16* __restrict__ e3, float* __restrict__ emax)
{
  int t = blockIdx.x * blockDim.x + threadIdx.x;
  if (t >= M1) return;
  const h16* p = e3 + (size_t)t * 256;
  float m = (float)p[0];
  for (int c = 1; c < 256; ++c) m = fmaxf(m, (float)p[c]);
  emax[t] = m;
}

// ===== softmax over K + keypoints = sum_k a*nbr (keypoints stored h16) =====
__global__ void softmax_kp_kernel(const float* __restrict__ emax, const float* __restrict__ nx,
                                  float* __restrict__ a, h16* __restrict__ keypoints)
{
  int bs = blockIdx.x * blockDim.x + threadIdx.x;
  if (bs >= BDIM * SPT) return;
  const size_t base = (size_t)bs * KNB;
  float mx = -PINF;
  for (int k = 0; k < KNB; ++k) mx = fmaxf(mx, emax[base + k]);
  float sum = 0.0f;
  for (int k = 0; k < KNB; ++k) { float e = __expf(emax[base + k] - mx); a[base + k] = e; sum += e; }
  float inv = 1.0f / sum;
  float kx = 0.f, ky = 0.f, kz = 0.f;
  for (int k = 0; k < KNB; ++k) {
    float w = a[base + k] * inv;
    a[base + k] = w;
    kx += nx[(base + k) * 3 + 0] * w;
    ky += nx[(base + k) * 3 + 1] * w;
    kz += nx[(base + k) * 3 + 2] * w;
  }
  keypoints[(size_t)bs * 3 + 0] = (h16)kx;
  keypoints[(size_t)bs * 3 + 1] = (h16)ky;
  keypoints[(size_t)bs * 3 + 2] = (h16)kz;
}

// ===== max over neighbor axis (h16): out[row, ooff+c] = max_k in[(row*Kk+k), c] =====
__global__ void maxk_kernel(const h16* __restrict__ in, h16* __restrict__ out,
                            int rows, int Kk, int C, int ostride, int ooff)
{
  int t = blockIdx.x * blockDim.x + threadIdx.x;
  if (t >= rows * C) return;
  int row = t / C, c = t % C;
  const h16* p = in + (size_t)row * Kk * C + c;
  float m = (float)p[0];
  for (int k = 1; k < Kk; ++k) m = fmaxf(m, (float)p[(size_t)k * C]);
  out[(size_t)row * ostride + ooff + c] = (h16)m;
}

// ===== edge features gf = [f_j - f_i, f_i, 0-pad], h16, row stride ldout =====
__global__ void graph_feature_kernel(const h16* __restrict__ feat, int ld, int C,
                                     const int* __restrict__ idx, h16* __restrict__ gf,
                                     int P, int GKk, int ldout, int total)
{
  int t = blockIdx.x * blockDim.x + threadIdx.x;
  if (t >= total) return;
  int c = t % ldout;
  int r = t / ldout;                // [0, B*P*GK)
  int bi = r / GKk;                 // [0, B*P)
  int b = bi / P, i = bi % P;
  float v = 0.0f;
  if (c < 2 * C) {
    if (c < C) {
      int j = idx[r];
      float fi = (float)feat[((size_t)b * P + i) * ld + c];
      float fj = (float)feat[((size_t)b * P + j) * ld + c];
      v = fj - fi;
    } else {
      v = (float)feat[((size_t)b * P + i) * ld + (c - C)];
    }
  }
  gf[(size_t)r * ldout + c] = (h16)v;
}

// ===== max+mean pool over points: h16 in, f32 out =====
__global__ void pool_kernel(const h16* __restrict__ x5, float* __restrict__ h, int P, int C)
{
  int t = blockIdx.x * blockDim.x + threadIdx.x;
  if (t >= BDIM * C) return;
  int b = t / C, c = t % C;
  const h16* p = x5 + (size_t)b * P * C + c;
  float mx = -PINF, sum = 0.0f;
  for (int i = 0; i < P; ++i) { float v = (float)p[(size_t)i * C]; mx = fmaxf(mx, v); sum += v; }
  h[(size_t)b * 2 * C + c]     = mx;
  h[(size_t)b * 2 * C + C + c] = sum / (float)P;
}

// ===== tiny dense layer (B=4 rows), f32 head =====
__global__ void small_linear_kernel(const float* __restrict__ X, const float* __restrict__ W,
                                    const float* __restrict__ bias, const float* __restrict__ gamma,
                                    const float* __restrict__ beta, float* __restrict__ Y,
                                    int rows, int K, int N, int act)
{
  int t = blockIdx.x * blockDim.x + threadIdx.x;
  if (t >= rows * N) return;
  int n = t % N, r = t / N;
  const float* x = X + (size_t)r * K;
  const float* w = W + (size_t)n * K;
  float s = 0.0f;
  for (int k = 0; k < K; ++k) s += x[k] * w[k];
  if (bias)  s += bias[n];
  if (gamma) s = s * (gamma[n] * rsqrtf(1.0f + 1e-5f)) + beta[n];
  Y[t] = apply_act(s, act);
}

// ===== concat + L2-normalize =====
__global__ void norm_out_kernel(const float* __restrict__ kf, const float* __restrict__ df,
                                float* __restrict__ out)
{
  __shared__ float ssum[256];
  int b = blockIdx.x, t = threadIdx.x;
  float v = (t < 128) ? kf[(size_t)b * 128 + t] : df[(size_t)b * 128 + (t - 128)];
  ssum[t] = v * v;
  __syncthreads();
  for (int off = 128; off > 0; off >>= 1) {
    if (t < off) ssum[t] += ssum[t + off];
    __syncthreads();
  }
  float inv = rsqrtf(ssum[0]);
  out[(size_t)b * 256 + t] = v * inv;
}

// =============================== host side ===================================
static void run_dgcnn(const h16* feat0, int C0, int ld0, const h16* W1, int K1, int ldg1,
                      const float* bn1g, const float* bn1b,
                      const h16* w2, const float* bn2g, const float* bn2b,
                      const h16* w3, const float* bn3g, const float* bn3b,
                      const h16* w4, const float* bn4g, const float* bn4b,
                      const h16* w5, const float* bn5g, const float* bn5b,
                      const float* lin1, const float* bn6g, const float* bn6b,
                      const float* lin2W, const float* lin2b,
                      const float* bn7g, const float* bn7b,
                      const float* lin3W, const float* lin3b,
                      int* idx20, h16* gf, h16* convout, h16* xcat, h16* x5,
                      float* hbuf, float* t1, float* t2, float* outvec, hipStream_t stream)
{
  const int P = SPT;
  struct Lay { const h16* feat; int ld; int C; const h16* W; int Kg; int ldg;
               const float* g; const float* b; int Cout; int choff; };
  Lay L[4] = {
    { feat0,      ld0, C0,  W1, K1,  ldg1, bn1g, bn1b,  64,   0 },
    { xcat + 0,   512, 64,  w2, 128, 128,  bn2g, bn2b,  64,  64 },
    { xcat + 64,  512, 64,  w3, 128, 128,  bn3g, bn3b, 128, 128 },
    { xcat + 128, 512, 128, w4, 256, 256,  bn4g, bn4b, 256, 256 },
  };
  for (int l = 0; l < 4; ++l) {
    size_t sh = (size_t)P * 4 + 64 * 4 + 64 * 4;
    knn_kernel<<<BDIM * P, 64, sh, stream>>>(L[l].feat, L[l].ld, L[l].C, P, GKNB, idx20);
    int tot = M2 * L[l].ldg;
    graph_feature_kernel<<<(tot + 255) / 256, 256, 0, stream>>>(
        L[l].feat, L[l].ld, L[l].C, idx20, gf, P, GKNB, L[l].ldg, tot);
    gemm_h16_nt<4><<<dim3(M2 / 16, L[l].Cout / 64), 32, 0, stream>>>(
        gf, L[l].W, nullptr, L[l].g, L[l].b, convout, M2, L[l].Kg, L[l].Cout, 2);
    int tot2 = BDIM * P * L[l].Cout;
    maxk_kernel<<<(tot2 + 255) / 256, 256, 0, stream>>>(
        convout, xcat, BDIM * P, GKNB, L[l].Cout, 512, L[l].choff);
  }
  gemm_h16_nt<4><<<dim3((BDIM * P) / 16, 1024 / 64), 32, 0, stream>>>(
      xcat, w5, nullptr, bn5g, bn5b, x5, BDIM * P, 512, 1024, 2);
  pool_kernel<<<(BDIM * 1024 + 255) / 256, 256, 0, stream>>>(x5, hbuf, P, 1024);
  small_linear_kernel<<<(BDIM * 512 + 127) / 128, 128, 0, stream>>>(
      hbuf, lin1, nullptr, bn6g, bn6b, t1, BDIM, 2048, 512, 2);
  small_linear_kernel<<<(BDIM * 256 + 127) / 128, 128, 0, stream>>>(
      t1, lin2W, lin2b, bn7g, bn7b, t2, BDIM, 512, 256, 2);
  small_linear_kernel<<<(BDIM * 128 + 127) / 128, 128, 0, stream>>>(
      t2, lin3W, lin3b, nullptr, nullptr, outvec, BDIM, 256, 128, 0);
}

extern "C" void kernel_launch(void* const* d_in, const int* in_sizes, int n_in,
                              void* d_out, int out_size, void* d_ws, size_t ws_size,
                              hipStream_t stream)
{
  (void)in_sizes; (void)n_in; (void)out_size; (void)ws_size;
#define FIN(i) ((const float*)d_in[(i)])
  const float* pc      = FIN(0);
  const float* d_w1    = FIN(1);  const float* d_bn1g = FIN(2);  const float* d_bn1b = FIN(3);
  const float* d_w2    = FIN(4);  const float* d_bn2g = FIN(5);  const float* d_bn2b = FIN(6);
  const float* d_w3    = FIN(7);  const float* d_bn3g = FIN(8);  const float* d_bn3b = FIN(9);
  // m_w1..m_w3 (indices 10..19) only feed `sigmas`, which never reaches the
  // output => dead code, skipped.
  const float* s_w1    = FIN(20); const float* s_bn1g = FIN(21); const float* s_bn1b = FIN(22);
  const float* s_w2    = FIN(23); const float* s_bn2g = FIN(24); const float* s_bn2b = FIN(25);
  const float* s_w3    = FIN(26); const float* s_bn3g = FIN(27); const float* s_bn3b = FIN(28);
  const float* s_w4    = FIN(29); const float* s_bn4g = FIN(30); const float* s_bn4b = FIN(31);
  const float* s_w5    = FIN(32); const float* s_bn5g = FIN(33); const float* s_bn5b = FIN(34);
  const float* dg_ws1  = FIN(35);
  const float* dg_wf1  = FIN(36);
  const float* dg_bn1g = FIN(37); const float* dg_bn1b = FIN(38);
  const float* dg_w2   = FIN(39); const float* dg_bn2g = FIN(40); const float* dg_bn2b = FIN(41);
  const float* dg_w3   = FIN(42); const float* dg_bn3g = FIN(43); const float* dg_bn3b = FIN(44);
  const float* dg_w4   = FIN(45); const float* dg_bn4g = FIN(46); const float* dg_bn4b = FIN(47);
  const float* dg_w5   = FIN(48); const float* dg_bn5g = FIN(49); const float* dg_bn5b = FIN(50);
  const float* lin1    = FIN(51); const float* bn6g   = FIN(52); const float* bn6b   = FIN(53);
  const float* lin2W   = FIN(54); const float* lin2b  = FIN(55);
  const float* bn7g    = FIN(56); const float* bn7b   = FIN(57);
  const float* lin3W   = FIN(58); const float* lin3b  = FIN(59);
  const int* sidx = (const int*)d_in[60];
  const int* dsel = (const int*)d_in[61];
  float* out = (float*)d_out;

  // ---------------- workspace arena (deterministic bump allocator) ----------
  char* wsb = (char*)d_ws;
  size_t off = 0;
  auto alloc = [&](size_t bytes) -> void* {
    void* p = wsb + off;
    off = (off + bytes + 255) & ~(size_t)255;
    return p;
  };
  int*   top128    = (int*)  alloc((size_t)BDIM * SPT * DILK * 4);
  float* nx        = (float*)alloc((size_t)M1 * 3 * 4);
  h16*   cluster   = (h16*)  alloc((size_t)M1 * 32 * 2);   // padded to K=32
  float* a         = (float*)alloc((size_t)M1 * 4);
  float* emaxb     = (float*)alloc((size_t)M1 * 4);
  h16*   keypoints = (h16*)  alloc((size_t)BDIM * SPT * 3 * 2);
  h16*   ym        = (h16*)  alloc((size_t)BDIM * SPT * 128 * 2);
  h16*   descb     = (h16*)  alloc((size_t)BDIM * SPT * 32 * 2);
  int*   idx20     = (int*)  alloc((size_t)BDIM * SPT * GKNB * 4);
  float* hbuf      = (float*)alloc((size_t)BDIM * 2048 * 4);
  float* t1        = (float*)alloc((size_t)BDIM * 512 * 4);
  float* t2        = (float*)alloc((size_t)BDIM * 256 * 4);
  float* kf        = (float*)alloc((size_t)BDIM * 128 * 4);
  float* df        = (float*)alloc((size_t)BDIM * 128 * 4);
  // h16 weight copies (converted once per launch; L2-resident afterwards)
  h16* d_w1h  = (h16*)alloc((size_t)64  * 32  * 2);
  h16* d_w2h  = (h16*)alloc((size_t)128 * 64  * 2);
  h16* d_w3h  = (h16*)alloc((size_t)256 * 128 * 2);
  h16* s_w1h  = (h16*)alloc((size_t)64  * 32  * 2);
  h16* s_w2h  = (h16*)alloc((size_t)128 * 64  * 2);
  h16* s_w3h  = (h16*)alloc((size_t)128 * 128 * 2);
  h16* s_w4h  = (h16*)alloc((size_t)128 * 512 * 2);
  h16* s_w5h  = (h16*)alloc((size_t)32  * 128 * 2);
  h16* dgws1h = (h16*)alloc((size_t)64  * 32  * 2);
  h16* dgwf1h = (h16*)alloc((size_t)64  * 64  * 2);
  h16* dgw2h  = (h16*)alloc((size_t)64  * 128 * 2);
  h16* dgw3h  = (h16*)alloc((size_t)128 * 128 * 2);
  h16* dgw4h  = (h16*)alloc((size_t)256 * 256 * 2);
  h16* dgw5h  = (h16*)alloc((size_t)1024* 512 * 2);
  h16* R1 = (h16*)alloc((size_t)M1 * 64  * 2);  // e1 -> y1 -> y5
  h16* R2 = (h16*)alloc((size_t)M1 * 128 * 2);  // e2 -> y2 -> y4
  h16* R3 = (h16*)alloc((size_t)M1 * 256 * 2);  // e3 -> dgcnn scratch
  h16* R4 = (h16*)alloc((size_t)M1 * 128 * 2);  // y3
  // dgcnn scratch aliases R3 (e3 dead once y4 is produced)
  h16* gf      = R3;
  h16* convout = R3 + (size_t)M2 * 256;
  h16* xcat    = convout + (size_t)M2 * 256;
  h16* x5      = xcat + (size_t)BDIM * SPT * 512;

  // ---------------- 0) convert weights to h16 (pad K to %32) ----------------
  struct WCvt { const float* src; h16* dst; int N, K, Kp; };
  const WCvt wc[14] = {
    { d_w1,   d_w1h,  64,  4,   32  }, { d_w2,  d_w2h,  128, 64,  64  },
    { d_w3,   d_w3h,  256, 128, 128 }, { s_w1,  s_w1h,  64,  4,   32  },
    { s_w2,   s_w2h,  128, 64,  64  }, { s_w3,  s_w3h,  128, 128, 128 },
    { s_w4,   s_w4h,  128, 512, 512 }, { s_w5,  s_w5h,  32,  128, 128 },
    { dg_ws1, dgws1h, 64,  6,   32  }, { dg_wf1,dgwf1h, 64,  64,  64  },
    { dg_w2,  dgw2h,  64,  128, 128 }, { dg_w3, dgw3h,  128, 128, 128 },
    { dg_w4,  dgw4h,  256, 256, 256 }, { dg_w5, dgw5h,  1024,512, 512 },
  };
  for (int i = 0; i < 14; ++i)
    cvt_weights_kernel<<<(wc[i].N * wc[i].Kp + 255) / 256, 256, 0, stream>>>(
        wc[i].src, wc[i].dst, wc[i].N, wc[i].K, wc[i].Kp);

  // ---------------- 1) neighborhood selection (LDS-resident top-128) --------
  {
    size_t sh = (size_t)NPT * 4 + 256 * 4 + 256 * 4;   // 67.5 KB < 320 KB/WGP
    topk128_kernel<<<BDIM * SPT, 256, sh, stream>>>(pc, sidx, top128);
  }
  gather_cluster_kernel<<<(M1 + 255) / 256, 256, 0, stream>>>(pc, sidx, dsel, top128, nx, cluster);

  // ---------------- 2) cluster MLP towers (all WMMA, all-h16 data path) -----
  gemm_h16_nt<4><<<dim3(M1/16, 1), 32, 0, stream>>>(cluster, d_w1h, nullptr, d_bn1g, d_bn1b, R1, M1,  32,  64, 1);
  gemm_h16_nt<4><<<dim3(M1/16, 2), 32, 0, stream>>>(R1,      d_w2h, nullptr, d_bn2g, d_bn2b, R2, M1,  64, 128, 1);
  gemm_h16_nt<4><<<dim3(M1/16, 4), 32, 0, stream>>>(R2,      d_w3h, nullptr, d_bn3g, d_bn3b, R3, M1, 128, 256, 1);
  emax_kernel<<<(M1 + 255) / 256, 256, 0, stream>>>(R3, emaxb);
  softmax_kp_kernel<<<(BDIM * SPT + 255) / 256, 256, 0, stream>>>(emaxb, nx, a, keypoints);

  gemm_h16_nt<4><<<dim3(M1/16, 1), 32, 0, stream>>>(cluster, s_w1h, nullptr, s_bn1g, s_bn1b, R1, M1,  32,  64, 1);
  gemm_h16_nt<4><<<dim3(M1/16, 2), 32, 0, stream>>>(R1,      s_w2h, nullptr, s_bn2g, s_bn2b, R2, M1,  64, 128, 1);
  gemm_h16_nt<4><<<dim3(M1/16, 2), 32, 0, stream>>>(R2,      s_w3h, nullptr, s_bn3g, s_bn3b, R4, M1, 128, 128, 1);
  maxk_kernel<<<(BDIM * SPT * 128 + 255) / 256, 256, 0, stream>>>(R4, ym, BDIM * SPT, KNB, 128, 128, 0);
  gemm_y4_kernel<<<dim3(M1/16, 2), 32, 0, stream>>>(ym, R4, R3, a, s_w4h, s_bn4g, s_bn4b, R2, M1);
  gemm_h16_nt<2><<<dim3(M1/16, 1), 32, 0, stream>>>(R2, s_w5h, nullptr, s_bn5g, s_bn5b, R1, M1, 128, 32, 1);
  maxk_kernel<<<(BDIM * SPT * 32 + 255) / 256, 256, 0, stream>>>(R1, descb, BDIM * SPT, KNB, 32, 32, 0);

  // ---------------- 3) two DGCNN towers -------------------------------------
  run_dgcnn(keypoints, 3, 3, dgws1h, 32, 32, dg_bn1g, dg_bn1b,
            dgw2h, dg_bn2g, dg_bn2b, dgw3h, dg_bn3g, dg_bn3b,
            dgw4h, dg_bn4g, dg_bn4b, dgw5h, dg_bn5g, dg_bn5b,
            lin1, bn6g, bn6b, lin2W, lin2b, bn7g, bn7b, lin3W, lin3b,
            idx20, gf, convout, xcat, x5, hbuf, t1, t2, kf, stream);
  run_dgcnn(descb, 32, 32, dgwf1h, 64, 64, dg_bn1g, dg_bn1b,
            dgw2h, dg_bn2g, dg_bn2b, dgw3h, dg_bn3g, dg_bn3b,
            dgw4h, dg_bn4g, dg_bn4b, dgw5h, dg_bn5g, dg_bn5b,
            lin1, bn6g, bn6b, lin2W, lin2b, bn7g, bn7b, lin3W, lin3b,
            idx20, gf, convout, xcat, x5, hbuf, t1, t2, df, stream);

  // ---------------- 4) concat + L2 normalize --------------------------------
  norm_out_kernel<<<BDIM, 256, 0, stream>>>(kf, df, out);
#undef FIN
}
